// CO_FOG_61211873903194
// MI455X (gfx1250) — matrix operand order repaired and checked by reading, not compiled, not run on hardware
//
#include <hip/hip_runtime.h>

// ---------------------------------------------------------------------------
// MI455X (gfx1250) fused GNN-ensemble kernel, round 4.
// ~8 GFLOP of GEMM vs ~21MB input -> compute bound -> f16 WMMA pipe
// (v_wmma_f32_16x16x32_f16).  Changes vs round 3:
//  * class loop processes TWO classes per iteration through ping-pong
//    4KB buffers (t1a/t1b/t2a/t2b in the dead cH region): barrier count
//    halves (20 -> 10) and per-barrier WMMA work doubles.
//  * stage-6 output dot parallelized over 128 threads + LDS reduction.
// Everything else unchanged: TB=16 (one class == one full 16-row M-tile,
// zero padding anywhere), 64,640B LDS arena, 4 workgroups / WGP.
// ---------------------------------------------------------------------------

typedef __attribute__((ext_vector_type(16))) _Float16 v16h;
typedef __attribute__((ext_vector_type(8)))  _Float16 v8h;
typedef __attribute__((ext_vector_type(8)))  float    v8f;

#define TB    16           // batches per workgroup
#define NCLS  10
#define FEAT  64
#define HID   128
#define BATCH 8192
#define ROWS  (TB * NCLS)  // 160 activation rows (r = class*16 + batch)

// packed-weight layout in d_ws (units: halves). tile = 512 halves (32Kx16N),
// lane-major: tilebase + lane*16 + e;  lane<16 -> (n=lane, k=0..7,16..23),
// lane>=16 -> (n=lane-16, k=8..15,24..31).
#define PK_W1A 0
#define PK_W1B 4096
#define PK_G1  8192
#define PK_G2  16384
#define PK_D1  32768
#define PK_TOT 196608      // 384 tiles * 512 halves = 384 KB as f16

// ---------------------------------------------------------------------------
__global__ void __launch_bounds__(256)
pack_weights(const float* __restrict__ W1, const float* __restrict__ G1,
             const float* __restrict__ G2, const float* __restrict__ D1,
             _Float16* __restrict__ out)
{
    int p = blockIdx.x * blockDim.x + threadIdx.x;
    if (p >= PK_TOT) return;
    int tile = p >> 9;               // /512
    int q    = p & 511;
    int lane = q >> 4, e = q & 15;
    int n0 = lane & 15;
    int k0 = ((lane >> 4) << 3) + (e < 8 ? e : e + 8); // 0..31
    float v;
    if (tile < 8) {                  // W1a: rows 0..63 of W1[128][64]
        int kt = tile >> 2, nt = tile & 3;
        v = W1[(kt * 32 + k0) * 64 + nt * 16 + n0];
    } else if (tile < 16) {          // W1b: rows 64..127 of W1
        int t = tile - 8; int kt = t >> 2, nt = t & 3;
        v = W1[(64 + kt * 32 + k0) * 64 + nt * 16 + n0];
    } else if (tile < 32) {          // G1 [64][128]
        int t = tile - 16; int kt = t >> 3, nt = t & 7;
        v = G1[(kt * 32 + k0) * 128 + nt * 16 + n0];
    } else if (tile < 64) {          // G2 [128][128]
        int t = tile - 32; int kt = t >> 3, nt = t & 7;
        v = G2[(kt * 32 + k0) * 128 + nt * 16 + n0];
    } else {                         // D1 [1280][128]
        int t = tile - 64; int kt = t >> 3, nt = t & 7;
        v = D1[(kt * 32 + k0) * 128 + nt * 16 + n0];
    }
    out[p] = (_Float16)v;
}

// ---------------------------------------------------------------------------
// WMMA operand helpers (ISA 7.12.2, wave32).  Branch-free, no padding rows.
// ---------------------------------------------------------------------------
__device__ __forceinline__ v16h load_a(const _Float16* buf, int pitch,
                                       int mbase, int kbase)
{
    int l = threadIdx.x & 31;
    const _Float16* p = buf + (mbase + (l & 15)) * pitch
                            + kbase + ((l >> 4) << 3);
    v8h lo = *(const v8h*)(p);
    v8h hi = *(const v8h*)(p + 16);
    return __builtin_shufflevector(lo, hi, 0,1,2,3,4,5,6,7,8,9,10,11,12,13,14,15);
}

__device__ __forceinline__ void store_d_f16(_Float16* buf, int pitch, int mbase,
                                            int nbase, v8f d, const float* bias,
                                            bool relu)
{
    int l  = threadIdx.x & 31;
    int n  = nbase + (l & 15);
    int mh = (l >> 4) << 3;
    float bv = bias ? bias[n] : 0.0f;
#pragma unroll
    for (int v = 0; v < 8; ++v) {
        float x = d[v] + bv;
        if (relu) x = x > 0.0f ? x : 0.0f;
        buf[(mbase + mh + v) * pitch + n] = (_Float16)x;
    }
}

#define WMMA(A, B, C) \
    __builtin_amdgcn_wmma_f32_16x16x32_f16(false, (A), false, (B), (short)0, (C), false, false)

// ---------------------------------------------------------------------------
// Fused kernel: one WG = 16 batches, 256 threads (8 wave32).
// LDS arena (64,640 B), aliased:
//   [0     ,20480) nodeH f16[160][64]     -> later zF f32[16][128] + partials
//   [20480 ,40960) aH    f16[160][64]     -> later xH f16[160][64]
//   [40960 ,61440) cH    f16[160][64]     -> later t1a/t1b/t2a/t2b (4x4KB)
//   [61440 ,64640) adjH  f16[16][10][10]
// ---------------------------------------------------------------------------
__global__ void __launch_bounds__(256)
fused_gnn(const float* __restrict__ hidden,
          const float* __restrict__ b1, const float* __restrict__ W2,
          const float* __restrict__ b2, const float* __restrict__ g1,
          const float* __restrict__ g2, const float* __restrict__ d1,
          const float* __restrict__ D2, const float* __restrict__ d2,
          const _Float16* __restrict__ pk, float* __restrict__ out)
{
    __shared__ __align__(16) char smem[64640];
    _Float16* nodeH = (_Float16*)(smem);
    _Float16* aH    = (_Float16*)(smem + 20480);
    _Float16* cH    = (_Float16*)(smem + 40960);
    _Float16* adjH  = (_Float16*)(smem + 61440);
    _Float16* xH    = (_Float16*)(smem + 20480);
    _Float16* t1a   = (_Float16*)(smem + 40960);
    _Float16* t1b   = (_Float16*)(smem + 45056);
    _Float16* t2a   = (_Float16*)(smem + 49152);
    _Float16* t2b   = (_Float16*)(smem + 53248);
    float*    zF    = (float*)(smem);             // [16][128] = 8192B
    float*    part  = (float*)(smem + 8192);      // [128] stage-6 partials

    const int tid  = threadIdx.x;
    const int wave = tid >> 5;
    const int loff = (tid & 31) * 16;      // B-tile lane offset (halves)
    const int b0   = blockIdx.x * TB;

    // ---- Stage 1: load node tile -> nodeH (f16), r = class*16 + batch -----
    for (int e = tid; e < ROWS * FEAT; e += 256) {
        int r = e >> 6, f = e & 63;
        int i = r >> 4, b = r & 15;
        nodeH[e] = (_Float16)hidden[((size_t)i * BATCH + b0 + b) * FEAT + f];
    }
    __syncthreads();

    // ---- Stage 2: projections a = node@W1a + b1 ; c = node@W1b (WMMA) -----
    for (int job = wave; job < 80; job += 8) {
        bool isA = job < 40;
        int  t   = isA ? job : job - 40;
        int  mt  = t >> 2, nt = t & 3;
        const _Float16* wp = pk + (isA ? PK_W1A : PK_W1B);
        v8f acc = {};
#pragma unroll
        for (int kt = 0; kt < 2; ++kt) {
            v16h A = load_a(nodeH, FEAT, mt * 16, kt * 32);
            v16h B = *(const v16h*)(wp + (kt * 4 + nt) * 512 + loff);
            acc = WMMA(A, B, acc);
        }
        store_d_f16(isA ? aH : cH, FEAT, mt * 16, nt * 16, acc,
                    isA ? b1 : nullptr, false);
    }
    __syncthreads();

    // ---- Stage 3a: pairwise adjacency, one (b,i,j) pair per thread --------
    const float B2 = b2[0];
    for (int p = tid; p < TB * NCLS * NCLS; p += 256) {
        int b  = p & 15;
        int ij = p >> 4;
        int j  = ij % NCLS, i = ij / NCLS;
        const _Float16* arow = aH + (i * TB + b) * FEAT;
        const _Float16* crow = cH + (j * TB + b) * FEAT;
        float s = B2;
        for (int k = 0; k < FEAT; ++k) {
            float h = (float)arow[k] + (float)crow[k];
            h = h > 0.0f ? h : 0.0f;
            s += h * W2[k];
        }
        float av = (i == j) ? 1.0f : 1.0f / (1.0f + __expf(-s));
        adjH[(b * NCLS + i) * NCLS + j] = (_Float16)av;
    }
    __syncthreads();

    // ---- Stage 3b: x = adj@node + node  -> xH (f16, overwrites aH) --------
    for (int e = tid; e < ROWS * FEAT; e += 256) {
        int r = e >> 6, f = e & 63;
        int i = r >> 4, b = r & 15;
        float s = (float)nodeH[e];
        for (int j = 0; j < NCLS; ++j)
            s += (float)adjH[(b * NCLS + i) * NCLS + j]
               * (float)nodeH[(j * TB + b) * FEAT + f];
        xH[e] = (_Float16)s;
    }
    __syncthreads();

    // ---- Stages 4a+4b+5 fused, TWO classes per barrier interval. ----------
    // wave w owns output N-tile nt=w throughout; acc5 carried in VGPRs.
    {
        const int nt = wave;               // 8 waves x 8 N-tiles of HID=128
        v8f acc5 = {};
        for (int ip = 0; ip < NCLS; ip += 2) {
            // 4a: t1{a,b} = relu(x_{ip,ip+1} @ G1 + g1)
            v8f u0 = {}, u1 = {};
#pragma unroll
            for (int kt = 0; kt < 2; ++kt) {
                v16h B  = *(const v16h*)(pk + PK_G1 + (kt * 8 + nt) * 512 + loff);
                v16h A0 = load_a(xH, FEAT, ip * 16, kt * 32);
                v16h A1 = load_a(xH, FEAT, (ip + 1) * 16, kt * 32);
                u0 = WMMA(A0, B, u0);
                u1 = WMMA(A1, B, u1);
            }
            store_d_f16(t1a, HID, 0, nt * 16, u0, g1, true);
            store_d_f16(t1b, HID, 0, nt * 16, u1, g1, true);
            __syncthreads();

            // 4b: t2{a,b} = t1{a,b} @ G2 + g2
            v8f w0 = {}, w1 = {};
#pragma unroll
            for (int kt = 0; kt < 4; ++kt) {
                v16h B  = *(const v16h*)(pk + PK_G2 + (kt * 8 + nt) * 512 + loff);
                v16h A0 = load_a(t1a, HID, 0, kt * 32);
                v16h A1 = load_a(t1b, HID, 0, kt * 32);
                w0 = WMMA(A0, B, w0);
                w1 = WMMA(A1, B, w1);
            }
            store_d_f16(t2a, HID, 0, nt * 16, w0, g2, false);
            store_d_f16(t2b, HID, 0, nt * 16, w1, g2, false);
            __syncthreads();

            // 5-partial: acc5 += t2{a,b} @ D1[k-slices of classes ip, ip+1]
#pragma unroll
            for (int kt = 0; kt < 4; ++kt) {
                v16h A0 = load_a(t2a, HID, 0, kt * 32);
                v16h B0 = *(const v16h*)(pk + PK_D1
                                         + ((ip * 4 + kt) * 8 + nt) * 512 + loff);
                v16h A1 = load_a(t2b, HID, 0, kt * 32);
                v16h B1 = *(const v16h*)(pk + PK_D1
                                         + (((ip + 1) * 4 + kt) * 8 + nt) * 512 + loff);
                acc5 = WMMA(A0, B0, acc5);
                acc5 = WMMA(A1, B1, acc5);
            }
            // no barrier here: next iteration's post-4a barrier orders the
            // t1/t2 overwrites against this iteration's reads.
        }

        // z = relu(acc5 + d1) -> zF (full 16 valid batch rows)
        int l  = tid & 31;
        int n  = nt * 16 + (l & 15);
        int mh = (l >> 4) << 3;
        float dv = d1[n];
#pragma unroll
        for (int v = 0; v < 8; ++v) {
            float z = acc5[v] + dv;
            zF[(mh + v) * HID + n] = z > 0.0f ? z : 0.0f;
        }
    }
    __syncthreads();

    // ---- Stage 6: out[b] = z@D2 + d2 (128-thread partial dot + reduce) ----
    if (tid < 128) {
        int b = tid >> 3, l8 = tid & 7;
        float s = 0.0f;
        for (int h = l8 * 16; h < l8 * 16 + 16; ++h)
            s += zF[b * HID + h] * D2[h];
        part[tid] = s;
    }
    __syncthreads();
    if (tid < TB) {
        float s = d2[0];
#pragma unroll
        for (int q = 0; q < 8; ++q) s += part[tid * 8 + q];
        out[b0 + tid] = s;
    }
}

// ---------------------------------------------------------------------------
extern "C" void kernel_launch(void* const* d_in, const int* in_sizes, int n_in,
                              void* d_out, int out_size, void* d_ws, size_t ws_size,
                              hipStream_t stream)
{
    (void)in_sizes; (void)n_in; (void)out_size; (void)ws_size;
    const float* hidden = (const float*)d_in[1];   // [10,8192,64]
    const float* W1 = (const float*)d_in[2];
    const float* b1 = (const float*)d_in[3];
    const float* W2 = (const float*)d_in[4];
    const float* b2 = (const float*)d_in[5];
    const float* G1 = (const float*)d_in[6];
    const float* g1 = (const float*)d_in[7];
    const float* G2 = (const float*)d_in[8];
    const float* g2 = (const float*)d_in[9];
    const float* D1 = (const float*)d_in[10];
    const float* d1 = (const float*)d_in[11];
    const float* D2 = (const float*)d_in[12];
    const float* d2 = (const float*)d_in[13];

    _Float16* pk = (_Float16*)d_ws;   // 384 KB packed f16 weight tiles

    pack_weights<<<PK_TOT / 256, 256, 0, stream>>>(W1, G1, G2, D1, pk);
    fused_gnn<<<BATCH / TB, 256, 0, stream>>>(hidden, b1, W2, b2, g1, g2, d1,
                                              D2, d2, pk, (float*)d_out);
}
